// TemporalAttentionModule_8890582302865
// MI455X (gfx1250) — compile-verified, matrix operand order checked by reading, not run on hardware
//
#include <hip/hip_runtime.h>
#include <hip/hip_bf16.h>

// ---------------------------------------------------------------------------
// TemporalAttentionModule for gfx1250 (CDNA5, wave32, WMMA)
//   B=2, T=16, C=320, H=W=64  ->  8192 independent pixels, each a 16x320 seq.
//   5 GEMMs per pixel mapped onto v_wmma_f32_16x16x32_f16 (M=16=T, N-tiles=20,
//   K-chunks=10).  Weights pre-packed into WMMA B-fragment order (f16).
//   g-tile staged into LDS with global_load_async_to_lds_b128 (ASYNCcnt path).
// ---------------------------------------------------------------------------

typedef __attribute__((ext_vector_type(16))) _Float16 v16h;
typedef __attribute__((ext_vector_type(8)))  float    v8f;

#define TT     16          // tokens (T)
#define CDIM   320         // channels
#define NTILES 20          // C / 16
#define KCH    10          // C / 32
#define WAVES  5           // waves per block (160 threads)
#define NTPW   4           // N-tiles per wave (20 / 5)
#define HW     4096        // H*W
#define NPIX   8192        // B*H*W
#define LN1EPS 1e-5f
#define LN2EPS 1e-5f

union FragU { v16h v; float4 f[2]; };

// A-fragment (also B-fragment for row-major-K sources) from row-major f16 LDS:
// lane row = caller's `row`, two contiguous 16B chunks at kbase and kbase+16.
__device__ __forceinline__ v16h frag_rowmajor(const _Float16* buf, int row,
                                              int stride, int kbase) {
  FragU u;
  const _Float16* p = buf + row * stride + kbase;
  u.f[0] = *(const float4*)(p);
  u.f[1] = *(const float4*)(p + 16);
  return u.v;
}

// B-fragment from pre-packed global weights: 32B contiguous per lane.
__device__ __forceinline__ v16h frag_weight(const _Float16* wb, int mat, int nt,
                                            int kc, int lane) {
  const float4* p =
      (const float4*)(wb + ((((size_t)mat * NTILES + nt) * KCH + kc) * 32 + lane) * 16);
  FragU u;
  u.f[0] = p[0];
  u.f[1] = p[1];
  return u.v;
}

__device__ __forceinline__ v8f wmma_f16(v16h a, v16h b, v8f c) {
  return __builtin_amdgcn_wmma_f32_16x16x32_f16(false, a, false, b, (short)0, c,
                                                false, false);
}

// ---------------------------------------------------------------------------
// Kernel 1: pack wq/wk/wv/wf/w_po (f32 row-major [o][c]) into f16 B-fragment
// order: wb[mat][nt][kc][lane][elem], lane = n + 16*g, elem from K offset.
// ---------------------------------------------------------------------------
__global__ __launch_bounds__(256) void pack_weights(
    const float* __restrict__ wq, const float* __restrict__ wk,
    const float* __restrict__ wv, const float* __restrict__ wf,
    const float* __restrict__ wpo, _Float16* __restrict__ wb) {
  int i = blockIdx.x * 256 + threadIdx.x;
  if (i >= 5 * CDIM * CDIM) return;
  int m   = i / (CDIM * CDIM);
  int rem = i % (CDIM * CDIM);
  int o = rem / CDIM, c = rem % CDIM;
  const float* W = (m == 0) ? wq : (m == 1) ? wk : (m == 2) ? wv
                   : (m == 3) ? wf : wpo;
  float v = W[o * CDIM + c];
  int nt = o >> 4, n = o & 15;
  int kc = c >> 5, kk = c & 31;
  int g, e;
  if (kk < 16) { g = kk >> 3; e = kk & 7; }
  else         { int k2 = kk - 16; g = k2 >> 3; e = 8 + (k2 & 7); }
  int lane = n + 16 * g;
  size_t dst = (((size_t)m * NTILES + nt) * KCH + kc) * 32 * 16 + lane * 16 + e;
  wb[dst] = (_Float16)v;
}

// ---------------------------------------------------------------------------
// Kernel 2: x (BT, C, HW) f32  ->  gbuf (pixel, t, c) f16 via LDS tile.
// Coalesced reads over hw; 16B-granular writes per pixel row.
// ---------------------------------------------------------------------------
__global__ __launch_bounds__(256) void transpose_in(const float* __restrict__ x,
                                                    _Float16* __restrict__ gbuf) {
  __shared__ _Float16 tile[8][256];
  int blk = blockIdx.x;                 // 32 * 40 * 16 blocks
  int bt  = blk / 640;
  int rem = blk % 640;
  int cT  = rem / 16;                   // 40 c-tiles of 8
  int hwT = rem % 16;                   // 16 hw-tiles of 256
  int tid = threadIdx.x;
  int b = bt >> 4, t = bt & 15;
  int c0 = cT * 8, hw0 = hwT * 256;
  for (int cc = 0; cc < 8; ++cc) {
    float v = x[((size_t)bt * CDIM + c0 + cc) * HW + hw0 + tid];
    tile[cc][tid] = (_Float16)v;
  }
  __syncthreads();
  union { float4 f; _Float16 h[8]; } u;
  for (int cc = 0; cc < 8; ++cc) u.h[cc] = tile[cc][tid];
  size_t pix = (size_t)b * HW + hw0 + tid;
  *(float4*)(gbuf + (pix * TT + t) * CDIM + c0) = u.f;
}

// ---------------------------------------------------------------------------
// LayerNorm over C=320 for 16 rows, cooperative across 160 threads.
// ---------------------------------------------------------------------------
__device__ __forceinline__ void ln_block(const _Float16* IN, _Float16* OUT,
                                         const float* sc, const float* bi,
                                         float eps, float* sRed, float* sRed2,
                                         float* sMean, float* sRstd, int tid) {
  int row = tid & 15;
  int part = tid >> 4;                  // 0..9
  float s = 0.f, s2 = 0.f;
  for (int c = 0; c < 32; ++c) {
    float v = (float)IN[row * CDIM + part * 32 + c];
    s += v;
    s2 += v * v;
  }
  sRed[row * KCH + part]  = s;
  sRed2[row * KCH + part] = s2;
  __syncthreads();
  if (tid < 16) {
    float ts = 0.f, ts2 = 0.f;
    for (int p = 0; p < KCH; ++p) { ts += sRed[tid * KCH + p]; ts2 += sRed2[tid * KCH + p]; }
    float mean = ts * (1.f / CDIM);
    float var  = ts2 * (1.f / CDIM) - mean * mean;
    sMean[tid] = mean;
    sRstd[tid] = rsqrtf(var + eps);
  }
  __syncthreads();
  for (int i = 0; i < 32; ++i) {
    int e = tid * 32 + i;               // 160*32 == 5120
    int r = e / CDIM, c = e % CDIM;
    float v = (float)IN[r * CDIM + c];
    OUT[r * CDIM + c] = (_Float16)((v - sMean[r]) * sRstd[r] * sc[c] + bi[c]);
  }
}

// ---------------------------------------------------------------------------
// Kernel 3: one block per pixel. 160 threads = 5 wave32, 4 N-tiles per wave.
// ---------------------------------------------------------------------------
__global__ __launch_bounds__(160) void temporal_attn(
    const _Float16* __restrict__ gbuf, const _Float16* __restrict__ wb,
    const float* __restrict__ x, const float* __restrict__ bq,
    const float* __restrict__ bk, const float* __restrict__ bv,
    const float* __restrict__ ln1s, const float* __restrict__ ln1b,
    const float* __restrict__ bff, const float* __restrict__ ln2s,
    const float* __restrict__ ln2b, const float* __restrict__ bpo,
    float* __restrict__ out) {
  __shared__ _Float16 sG[TT * CDIM];     // g, later a2
  __shared__ _Float16 sQ[TT * CDIM];     // q, later h = attn+g
  __shared__ _Float16 sK[TT * CDIM];     // k, later n1
  __shared__ _Float16 sVt[CDIM * 32];    // v^T (padded K), later h2 row-major
  __shared__ _Float16 sP[TT * 32];       // softmax probs, zero-padded cols 16..31
  __shared__ float sRed[TT * KCH], sRed2[TT * KCH], sMean[TT], sRstd[TT];

  const int tid   = threadIdx.x;
  const int lane  = tid & 31;
  const int wave  = tid >> 5;            // 0..4
  const int half_ = lane >> 4;           // K-group
  const int lr    = lane & 15;           // row (A) / col (B,D)
  const int pix   = blockIdx.x;
  const int b     = pix >> 12;
  const int hw    = pix & (HW - 1);
  const float qscale = 0.05590169943749474f;   // 1/sqrt(320)

  // ---- stage g (16x320 f16, 10KB) via async global->LDS (ASYNCcnt) --------
  {
    uint32_t lds_base = (uint32_t)(uintptr_t)(void*)sG;   // LDS aperture: low 32b
    const char* gsrc = (const char*)(gbuf + (size_t)pix * TT * CDIM);
#pragma unroll
    for (int i = 0; i < 4; ++i) {                         // 640 x 16B chunks
      int off = (i * 160 + tid) * 16;
      uint32_t lds_off = lds_base + (uint32_t)off;
      uint64_t gaddr   = (uint64_t)(uintptr_t)(gsrc + off);
      asm volatile("global_load_async_to_lds_b128 %0, %1, off"
                   :: "v"(lds_off), "v"(gaddr)
                   : "memory");
    }
    // overlap: zero pad regions while the async copy is in flight
    for (int i = tid; i < CDIM * 32; i += 160) sVt[i] = (_Float16)0.f;
    for (int i = tid; i < TT * 32; i += 160) sP[i] = (_Float16)0.f;
    asm volatile("s_wait_asynccnt 0x0" ::: "memory");
  }
  __syncthreads();

  const v8f vzero = {};

  // ---- q / k / v GEMMs -----------------------------------------------------
  for (int m = 0; m < 3; ++m) {
    v8f acc[NTPW];
    for (int j = 0; j < NTPW; ++j) acc[j] = vzero;
    for (int kc = 0; kc < KCH; ++kc) {
      v16h a = frag_rowmajor(sG, lr, CDIM, kc * 32 + half_ * 8);
      if (kc + 1 < KCH)
        __builtin_prefetch(wb + ((((size_t)m * NTILES + wave * NTPW) * KCH + kc + 1) * 32) * 16, 0, 3);
      for (int j = 0; j < NTPW; ++j) {
        int nt = wave * NTPW + j;
        v16h bf = frag_weight(wb, m, nt, kc, lane);
        acc[j] = wmma_f16(a, bf, acc[j]);
      }
    }
    const float* bias = (m == 0) ? bq : (m == 1) ? bk : bv;
    for (int j = 0; j < NTPW; ++j) {
      int col = (wave * NTPW + j) * 16 + lr;
      float bb = bias[col];
      for (int r = 0; r < 8; ++r) {
        int t = half_ * 8 + r;
        float v = acc[j][r] + bb;
        if (m == 0)      sQ[t * CDIM + col] = (_Float16)(v * qscale);
        else if (m == 1) sK[t * CDIM + col] = (_Float16)v;
        else             sVt[col * 32 + t]  = (_Float16)v;   // transposed, pad zeros
      }
    }
  }
  __syncthreads();

  // ---- scores = (q*scale) @ k^T, softmax over s (wave 0 only) --------------
  if (wave == 0) {
    v8f sacc = vzero;
    for (int kc = 0; kc < KCH; ++kc) {
      v16h a  = frag_rowmajor(sQ, lr, CDIM, kc * 32 + half_ * 8);   // A: row t
      v16h bf = frag_rowmajor(sK, lr, CDIM, kc * 32 + half_ * 8);   // B: col s
      sacc = wmma_f16(a, bf, sacc);
    }
    // D layout: lane holds col s = lr, rows t = 8*half_ + r
    for (int r = 0; r < 8; ++r) {
      float v = sacc[r];
      float mx = v;
      for (int off = 8; off >= 1; off >>= 1) mx = fmaxf(mx, __shfl_xor(mx, off, 16));
      float e = __expf(v - mx);
      float sm = e;
      for (int off = 8; off >= 1; off >>= 1) sm += __shfl_xor(sm, off, 16);
      int t = half_ * 8 + r;
      sP[t * 32 + lr] = (_Float16)(e / sm);
    }
  }
  __syncthreads();

  // ---- attn_out = P @ V, residual h = attn_out + g -> sQ -------------------
  {
    v8f aacc[NTPW];
    for (int j = 0; j < NTPW; ++j) aacc[j] = vzero;
    v16h pa = frag_rowmajor(sP, lr, 32, half_ * 8);   // K=32 (s padded)
    for (int j = 0; j < NTPW; ++j) {
      int c0 = (wave * NTPW + j) * 16;
      v16h vb = frag_rowmajor(sVt, c0 + lr, 32, half_ * 8);  // B col c, contiguous s
      aacc[j] = wmma_f16(pa, vb, aacc[j]);
    }
    for (int j = 0; j < NTPW; ++j) {
      int col = (wave * NTPW + j) * 16 + lr;
      for (int r = 0; r < 8; ++r) {
        int t = half_ * 8 + r;
        float h = aacc[j][r] + (float)sG[t * CDIM + col];
        sQ[t * CDIM + col] = (_Float16)h;
      }
    }
  }
  __syncthreads();

  // ---- n1 = LN(h) -> sK ----------------------------------------------------
  ln_block(sQ, sK, ln1s, ln1b, LN1EPS, sRed, sRed2, sMean, sRstd, tid);
  __syncthreads();

  // ---- ffn: f = n1 @ wf^T + bf ; h2 = n1 + f -> sVt (row-major reuse) ------
  {
    _Float16* sH2 = sVt;
    v8f facc[NTPW];
    for (int j = 0; j < NTPW; ++j) facc[j] = vzero;
    for (int kc = 0; kc < KCH; ++kc) {
      v16h a = frag_rowmajor(sK, lr, CDIM, kc * 32 + half_ * 8);
      for (int j = 0; j < NTPW; ++j) {
        int nt = wave * NTPW + j;
        v16h bf = frag_weight(wb, 3, nt, kc, lane);
        facc[j] = wmma_f16(a, bf, facc[j]);
      }
    }
    for (int j = 0; j < NTPW; ++j) {
      int col = (wave * NTPW + j) * 16 + lr;
      float bb = bff[col];
      for (int r = 0; r < 8; ++r) {
        int t = half_ * 8 + r;
        float h2 = facc[j][r] + bb + (float)sK[t * CDIM + col];
        sH2[t * CDIM + col] = (_Float16)h2;
      }
    }
  }
  __syncthreads();

  // ---- a2 = LN(h2) -> sG ---------------------------------------------------
  ln_block(sVt, sG, ln2s, ln2b, LN2EPS, sRed, sRed2, sMean, sRstd, tid);
  __syncthreads();

  // ---- out = a2 @ w_po^T + b_po + x  (scatter back to (BT,C,HW)) -----------
  {
    v8f pacc[NTPW];
    for (int j = 0; j < NTPW; ++j) pacc[j] = vzero;
    for (int kc = 0; kc < KCH; ++kc) {
      v16h a = frag_rowmajor(sG, lr, CDIM, kc * 32 + half_ * 8);
      for (int j = 0; j < NTPW; ++j) {
        int nt = wave * NTPW + j;
        v16h bf = frag_weight(wb, 4, nt, kc, lane);
        pacc[j] = wmma_f16(a, bf, pacc[j]);
      }
    }
    for (int j = 0; j < NTPW; ++j) {
      int o = (wave * NTPW + j) * 16 + lr;
      float bb = bpo[o];
      for (int r = 0; r < 8; ++r) {
        int t = half_ * 8 + r;
        size_t idx = ((size_t)(b * TT + t) * CDIM + o) * HW + hw;
        out[idx] = pacc[j][r] + bb + x[idx];
      }
    }
  }
}

// ---------------------------------------------------------------------------
// Launch: pack weights -> transpose x -> fused per-pixel attention.
// Workspace: [0, 1MB)  = packed weights (1,024,000 B)
//            [1MB, ..) = gbuf f16 (8192*16*320*2 = 83,886,080 B)
// ---------------------------------------------------------------------------
extern "C" void kernel_launch(void* const* d_in, const int* in_sizes, int n_in,
                              void* d_out, int out_size, void* d_ws, size_t ws_size,
                              hipStream_t stream) {
  const float* x    = (const float*)d_in[0];
  const float* wq   = (const float*)d_in[5];
  const float* bq   = (const float*)d_in[6];
  const float* wk   = (const float*)d_in[7];
  const float* bk   = (const float*)d_in[8];
  const float* wv   = (const float*)d_in[9];
  const float* bv   = (const float*)d_in[10];
  const float* ln1s = (const float*)d_in[11];
  const float* ln1b = (const float*)d_in[12];
  const float* wf   = (const float*)d_in[13];
  const float* bff  = (const float*)d_in[14];
  const float* ln2s = (const float*)d_in[15];
  const float* ln2b = (const float*)d_in[16];
  const float* wpo  = (const float*)d_in[17];
  const float* bpo  = (const float*)d_in[18];
  float* out = (float*)d_out;

  _Float16* wb   = (_Float16*)d_ws;
  _Float16* gbuf = (_Float16*)((char*)d_ws + (1u << 20));

  int wtotal = 5 * CDIM * CDIM;
  pack_weights<<<(wtotal + 255) / 256, 256, 0, stream>>>(wq, wk, wv, wf, wpo, wb);
  transpose_in<<<32 * 40 * 16, 256, 0, stream>>>(x, gbuf);
  temporal_attn<<<NPIX, 160, 0, stream>>>(gbuf, wb, x, bq, bk, bv, ln1s, ln1b,
                                          bff, ln2s, ln2b, bpo, out);
}
